// ONGNN_method_80625126080941
// MI455X (gfx1250) — matrix operand order repaired and checked by reading, not compiled, not run on hardware
//
#include <hip/hip_runtime.h>
#include <hip/hip_bf16.h>

typedef __attribute__((ext_vector_type(16))) _Float16 v16h;
typedef __attribute__((ext_vector_type(8)))  float    v8f;

#define N_NODES 100000
#define IN_C 128
#define HID 128
#define OUT_C 40
#define CHUNK 64
#define LN_EPS 1e-5f
#define ROW_TILES (N_NODES / 16)          // 6250 (exact)
#define WAVES_PER_BLOCK 4
#define BLOCK_THREADS 128
#define TILE_GROUPS ((ROW_TILES + WAVES_PER_BLOCK - 1) / WAVES_PER_BLOCK)  // 1563
#define GEMM_BLOCKS 512                   // persistent blocks; ~3 tile-groups each

// ---------------------------------------------------------------------------
// A-fragment (16x32, f16) from a row-major f32 matrix [rows x ld].
// A layout: lanes 0-15 -> M=lane (K base +0), lanes 16-31 -> M=lane-16
// (K base +16); halves 0..15 -> K = kb..kb+15 (matches B fragment).
// `scale` fuses the scatter-mean division for the m-matrix.
// ---------------------------------------------------------------------------
__device__ __forceinline__ v16h load_a_frag(const float* __restrict__ base, int ld,
                                            int row0, int k0, int lane, float scale) {
    int row = row0 + (lane & 15);
    int kb  = k0 + ((lane >> 4) << 4);           // +16 for upper half-wave
    const float* p = base + (size_t)row * ld + kb;
    float4 f0 = *(const float4*)(p);
    float4 f1 = *(const float4*)(p + 4);
    float4 f2 = *(const float4*)(p + 8);
    float4 f3 = *(const float4*)(p + 12);
    v16h a;
    a[0]  = (_Float16)(f0.x * scale); a[1]  = (_Float16)(f0.y * scale);
    a[2]  = (_Float16)(f0.z * scale); a[3]  = (_Float16)(f0.w * scale);
    a[4]  = (_Float16)(f1.x * scale); a[5]  = (_Float16)(f1.y * scale);
    a[6]  = (_Float16)(f1.z * scale); a[7]  = (_Float16)(f1.w * scale);
    a[8]  = (_Float16)(f2.x * scale); a[9]  = (_Float16)(f2.y * scale);
    a[10] = (_Float16)(f2.z * scale); a[11] = (_Float16)(f2.w * scale);
    a[12] = (_Float16)(f3.x * scale); a[13] = (_Float16)(f3.y * scale);
    a[14] = (_Float16)(f3.z * scale); a[15] = (_Float16)(f3.w * scale);
    return a;
}

// B-fragment (32x16, f16) from an LDS weight copy stored K-major: Wt[n*ldk + k].
// Lanes 0-15 -> N=lane (K base +0), lanes 16-31 -> N=lane-16 (K base +16);
// 16 consecutive halves along K => one aligned 32B LDS read pair.
__device__ __forceinline__ v16h load_b_frag(const _Float16* Wt, int ldk,
                                            int n0, int k0, int lane) {
    int n  = n0 + (lane & 15);
    int kb = k0 + ((lane >> 4) << 4);            // +16 for upper half-wave
    return *(const v16h*)(Wt + (size_t)n * ldk + kb);
}

// Prefetch one 16-row f32 tile (16*HID*4 = 8KB) into near caches: 2 x 4KB sweeps.
__device__ __forceinline__ void prefetch_tile(const float* __restrict__ base,
                                              int row0, int lane) {
    const char* p = (const char*)(base + (size_t)row0 * HID);
    __builtin_prefetch(p + lane * 128, 0, 1);
    __builtin_prefetch(p + 4096 + lane * 128, 0, 1);
}

// ---------------------------------------------------------------------------
// Kernel 1: h = LayerNorm(relu(x @ W_in + b_in)) ; one wave = 16 rows.
// Persistent blocks: weights staged in LDS once, then grid-stride tile loop.
// ---------------------------------------------------------------------------
__global__ void gemm_in_ln_kernel(const float* __restrict__ x,
                                  const float* __restrict__ W,    // [IN_C][HID]
                                  const float* __restrict__ bias,
                                  const float* __restrict__ g,
                                  const float* __restrict__ be,
                                  float* __restrict__ h_out) {
    __shared__ __align__(32) _Float16 Wt[HID * IN_C];   // [n][k], 32 KB
    int tid = threadIdx.x;
    for (int i = tid; i < HID * IN_C; i += BLOCK_THREADS) {
        int n = i >> 7, k = i & 127;
        Wt[i] = (_Float16)W[k * HID + n];
    }
    __syncthreads();

    int wave = tid >> 5, lane = tid & 31;
    int rowAdd = (lane >> 4) ? 8 : 0;

    for (int tg = blockIdx.x; tg < TILE_GROUPS; tg += gridDim.x) {
        int tile = tg * WAVES_PER_BLOCK + wave;
        if (tile >= ROW_TILES) continue;
        int row0 = tile * 16;

        // prefetch next iteration's tile while this one computes
        int ntile = (tg + gridDim.x) * WAVES_PER_BLOCK + wave;
        if (ntile < ROW_TILES) prefetch_tile(x, ntile * 16, lane);

        v8f acc[8] = {};
#pragma unroll
        for (int kt = 0; kt < 4; kt++) {
            v16h a = load_a_frag(x, IN_C, row0, kt * 32, lane, 1.0f);
#pragma unroll
            for (int nt = 0; nt < 8; nt++) {
                v16h b = load_b_frag(Wt, IN_C, nt * 16, kt * 32, lane);
                acc[nt] = __builtin_amdgcn_wmma_f32_16x16x32_f16(
                    false, a, false, b, (short)0, acc[nt], false, false);
            }
        }
        // bias + relu
#pragma unroll
        for (int nt = 0; nt < 8; nt++) {
            float bc = bias[nt * 16 + (lane & 15)];
#pragma unroll
            for (int r = 0; r < 8; r++) {
                float v = acc[nt][r] + bc;
                acc[nt][r] = v > 0.0f ? v : 0.0f;
            }
        }
        // LayerNorm over 128 cols of each row (row = r + 8*(lane>=16))
        float mu[8], rs[8];
#pragma unroll
        for (int r = 0; r < 8; r++) {
            float s = 0.0f, q = 0.0f;
#pragma unroll
            for (int nt = 0; nt < 8; nt++) { float v = acc[nt][r]; s += v; q += v * v; }
            for (int off = 1; off < 16; off <<= 1) {
                s += __shfl_xor(s, off, 32);
                q += __shfl_xor(q, off, 32);
            }
            float m = s * (1.0f / HID);
            float var = q * (1.0f / HID) - m * m;
            mu[r] = m;
            rs[r] = rsqrtf(var + LN_EPS);
        }
#pragma unroll
        for (int nt = 0; nt < 8; nt++) {
            int col = nt * 16 + (lane & 15);
            float gc = g[col], bc = be[col];
#pragma unroll
            for (int r = 0; r < 8; r++) {
                int row = row0 + r + rowAdd;
                h_out[(size_t)row * HID + col] = (acc[nt][r] - mu[r]) * rs[r] * gc + bc;
            }
        }
    }
}

// ---------------------------------------------------------------------------
// scatter_mean helpers
// ---------------------------------------------------------------------------
__global__ void zero_kernel(float* __restrict__ p, size_t n) {
    size_t i = (size_t)blockIdx.x * blockDim.x + threadIdx.x;
    size_t stride = (size_t)gridDim.x * blockDim.x;
    for (; i < n; i += stride) p[i] = 0.0f;
}

__global__ void degree_kernel(const int* __restrict__ ei, float* __restrict__ cnt, int E) {
    int e = blockIdx.x * blockDim.x + threadIdx.x;
    if (e < E) atomicAdd(&cnt[ei[E + e]], 1.0f);
}

// warp per edge: 32 lanes x float4 = 128 channels; src gathers hit L2 (h is 51MB)
__global__ void scatter_add_kernel(const int* __restrict__ ei,
                                   const float* __restrict__ h,
                                   float* __restrict__ m, int E) {
    int warp = (blockIdx.x * blockDim.x + threadIdx.x) >> 5;
    int lane = threadIdx.x & 31;
    if (warp >= E) return;
    int s = ei[warp];
    int d = ei[E + warp];
    float4 v = *(const float4*)(h + (size_t)s * HID + lane * 4);
    float* mp = m + (size_t)d * HID + lane * 4;
    atomicAdd(mp + 0, v.x);
    atomicAdd(mp + 1, v.y);
    atomicAdd(mp + 2, v.z);
    atomicAdd(mp + 3, v.w);
}

// ---------------------------------------------------------------------------
// ongnn_conv: t = softmax(concat(h, m/cnt) @ tmW + tmb); cumsum; gate with sig;
//             out = LN(h*s + m*(1-s)). One wave = 16 rows; persistent blocks.
// ---------------------------------------------------------------------------
__global__ void ongnn_conv_kernel(const float* __restrict__ h_in,
                                  const float* __restrict__ msum,
                                  const float* __restrict__ cnt,
                                  float* __restrict__ sig,          // [N][64] in/out
                                  const float* __restrict__ tmW,    // [256][64]
                                  const float* __restrict__ tmb,
                                  const float* __restrict__ lng,
                                  const float* __restrict__ lnb,
                                  float* __restrict__ h_out) {
    __shared__ __align__(32) _Float16 Wt[CHUNK * 2 * HID];              // [64][256], 32 KB
    __shared__ float rawLDS[WAVES_PER_BLOCK][16][CHUNK];                // 16 KB
    int tid = threadIdx.x;
    for (int i = tid; i < CHUNK * 2 * HID; i += BLOCK_THREADS) {
        int n = i >> 8, k = i & 255;
        Wt[i] = (_Float16)tmW[k * CHUNK + n];
    }
    __syncthreads();

    int wave = tid >> 5, lane = tid & 31;
    int rowAdd = (lane >> 4) ? 8 : 0;

    for (int tg = blockIdx.x; tg < TILE_GROUPS; tg += gridDim.x) {
        int tile = tg * WAVES_PER_BLOCK + wave;
        if (tile >= ROW_TILES) continue;
        int row0 = tile * 16;

        int ntile = (tg + gridDim.x) * WAVES_PER_BLOCK + wave;
        if (ntile < ROW_TILES) {
            prefetch_tile(h_in, ntile * 16, lane);
            prefetch_tile(msum, ntile * 16, lane);
        }

        int fragRow = row0 + (lane & 15);
        float invc = 1.0f / fmaxf(cnt[fragRow], 1.0f);

        v8f acc[4] = {};
#pragma unroll
        for (int kt = 0; kt < 8; kt++) {
            v16h a = (kt < 4)
                ? load_a_frag(h_in, HID, row0, kt * 32, lane, 1.0f)
                : load_a_frag(msum, HID, row0, (kt - 4) * 32, lane, invc);
#pragma unroll
            for (int nt = 0; nt < 4; nt++) {
                v16h b = load_b_frag(Wt, 2 * HID, nt * 16, kt * 32, lane);
                acc[nt] = __builtin_amdgcn_wmma_f32_16x16x32_f16(
                    false, a, false, b, (short)0, acc[nt], false, false);
            }
        }
        // + bias
#pragma unroll
        for (int nt = 0; nt < 4; nt++) {
            float bc = tmb[nt * 16 + (lane & 15)];
#pragma unroll
            for (int r = 0; r < 8; r++) acc[nt][r] += bc;
        }

        // softmax over 64 + cumsum + gate, per row
#pragma unroll
        for (int r = 0; r < 8; r++) {
            float mx = -3.0e38f;
#pragma unroll
            for (int nt = 0; nt < 4; nt++) mx = fmaxf(mx, acc[nt][r]);
            for (int off = 1; off < 16; off <<= 1) mx = fmaxf(mx, __shfl_xor(mx, off, 32));
            float s = 0.0f;
#pragma unroll
            for (int nt = 0; nt < 4; nt++) {
                float e = __expf(acc[nt][r] - mx);
                acc[nt][r] = e;
                s += e;
            }
            for (int off = 1; off < 16; off <<= 1) s += __shfl_xor(s, off, 32);
            float inv = 1.0f / s;

            int rowG = row0 + r + rowAdd;
            float run = 0.0f;
#pragma unroll
            for (int nt = 0; nt < 4; nt++) {
                float v = acc[nt][r] * inv;
                // inclusive prefix within 16-lane group
                for (int d = 1; d < 16; d <<= 1) {
                    float t = __shfl_up(v, d, 32);
                    if ((lane & 15) >= d) v += t;
                }
                float tot = __shfl(v, (lane & 16) + 15, 32);   // group total
                v += run;
                run += tot;
                int col = nt * 16 + (lane & 15);
                float prev  = sig[(size_t)rowG * CHUNK + col];
                float gated = prev + (1.0f - prev) * v;
                sig[(size_t)rowG * CHUNK + col] = gated;
                rawLDS[wave][r + rowAdd][col] = gated;
            }
        }

        // output: v = h*s + m*(1-s); LayerNorm per row. Same-wave LDS RAW -> hw waits.
        for (int r2 = 0; r2 < 16; r2++) {
            int row = row0 + r2;
            float ic = 1.0f / fmaxf(cnt[row], 1.0f);
            float4 h4 = *(const float4*)(h_in + (size_t)row * HID + lane * 4);
            float4 m4 = *(const float4*)(msum + (size_t)row * HID + lane * 4);
            m4.x *= ic; m4.y *= ic; m4.z *= ic; m4.w *= ic;
            float sA = rawLDS[wave][r2][lane * 2];        // cols 4l,4l+1 -> chunk 2l
            float sB = rawLDS[wave][r2][lane * 2 + 1];    // cols 4l+2,4l+3 -> chunk 2l+1
            float v0 = h4.x * sA + m4.x * (1.0f - sA);
            float v1 = h4.y * sA + m4.y * (1.0f - sA);
            float v2 = h4.z * sB + m4.z * (1.0f - sB);
            float v3 = h4.w * sB + m4.w * (1.0f - sB);
            float s = v0 + v1 + v2 + v3;
            float q = v0 * v0 + v1 * v1 + v2 * v2 + v3 * v3;
            for (int off = 1; off < 32; off <<= 1) {
                s += __shfl_xor(s, off, 32);
                q += __shfl_xor(q, off, 32);
            }
            float mu = s * (1.0f / HID);
            float var = q * (1.0f / HID) - mu * mu;
            float rsv = rsqrtf(var + LN_EPS);
            int c0 = lane * 4;
            float o0 = (v0 - mu) * rsv * lng[c0 + 0] + lnb[c0 + 0];
            float o1 = (v1 - mu) * rsv * lng[c0 + 1] + lnb[c0 + 1];
            float o2 = (v2 - mu) * rsv * lng[c0 + 2] + lnb[c0 + 2];
            float o3 = (v3 - mu) * rsv * lng[c0 + 3] + lnb[c0 + 3];
            float4 o = make_float4(o0, o1, o2, o3);
            *(float4*)(h_out + (size_t)row * HID + c0) = o;
        }
    }
}

// ---------------------------------------------------------------------------
// Kernel 4: out = h @ W_out + b_out   (40 cols padded to 48 = 3 tiles)
// ---------------------------------------------------------------------------
__global__ void gemm_out_kernel(const float* __restrict__ h,
                                const float* __restrict__ Wout,   // [HID][OUT_C]
                                const float* __restrict__ bout,
                                float* __restrict__ out) {
    __shared__ __align__(32) _Float16 Wt[48 * HID];   // [n][k], zero-padded, 12 KB
    int tid = threadIdx.x;
    for (int i = tid; i < 48 * HID; i += BLOCK_THREADS) {
        int n = i >> 7, k = i & 127;
        Wt[i] = (n < OUT_C) ? (_Float16)Wout[k * OUT_C + n] : (_Float16)0.0f;
    }
    __syncthreads();

    int wave = tid >> 5, lane = tid & 31;
    int rowAdd = (lane >> 4) ? 8 : 0;

    for (int tg = blockIdx.x; tg < TILE_GROUPS; tg += gridDim.x) {
        int tile = tg * WAVES_PER_BLOCK + wave;
        if (tile >= ROW_TILES) continue;
        int row0 = tile * 16;

        int ntile = (tg + gridDim.x) * WAVES_PER_BLOCK + wave;
        if (ntile < ROW_TILES) prefetch_tile(h, ntile * 16, lane);

        v8f acc[3] = {};
#pragma unroll
        for (int kt = 0; kt < 4; kt++) {
            v16h a = load_a_frag(h, HID, row0, kt * 32, lane, 1.0f);
#pragma unroll
            for (int nt = 0; nt < 3; nt++) {
                v16h b = load_b_frag(Wt, HID, nt * 16, kt * 32, lane);
                acc[nt] = __builtin_amdgcn_wmma_f32_16x16x32_f16(
                    false, a, false, b, (short)0, acc[nt], false, false);
            }
        }
#pragma unroll
        for (int nt = 0; nt < 3; nt++) {
            int col = nt * 16 + (lane & 15);
            if (col < OUT_C) {
                float bc = bout[col];
#pragma unroll
                for (int r = 0; r < 8; r++) {
                    int row = row0 + r + rowAdd;
                    out[(size_t)row * OUT_C + col] = acc[nt][r] + bc;
                }
            }
        }
    }
}

// ---------------------------------------------------------------------------
extern "C" void kernel_launch(void* const* d_in, const int* in_sizes, int n_in,
                              void* d_out, int out_size, void* d_ws, size_t ws_size,
                              hipStream_t stream) {
    const float* x     = (const float*)d_in[0];
    const int*   ei    = (const int*)d_in[1];
    const float* W_in  = (const float*)d_in[2];
    const float* b_in  = (const float*)d_in[3];
    const float* g_in  = (const float*)d_in[4];
    const float* be_in = (const float*)d_in[5];
    const float* tm_W0 = (const float*)d_in[6];
    const float* tm_b0 = (const float*)d_in[7];
    const float* ln_g0 = (const float*)d_in[8];
    const float* ln_b0 = (const float*)d_in[9];
    const float* tm_W1 = (const float*)d_in[10];
    const float* tm_b1 = (const float*)d_in[11];
    const float* ln_g1 = (const float*)d_in[12];
    const float* ln_b1 = (const float*)d_in[13];
    const float* W_out = (const float*)d_in[14];
    const float* b_out = (const float*)d_in[15];
    float* out = (float*)d_out;

    float* ws   = (float*)d_ws;
    float* h_a  = ws;                                    // N*128
    float* h_b  = h_a  + (size_t)N_NODES * HID;          // N*128
    float* msum = h_b  + (size_t)N_NODES * HID;          // N*128
    float* cnt  = msum + (size_t)N_NODES * HID;          // N
    float* sig  = cnt  + N_NODES;                        // N*64

    int E = in_sizes[1] / 2;
    int edgeBlocks = (E + 255) / 256;
    int scatBlocks = (E * 32 + 255) / 256;

    // cnt and sig are contiguous: zero both in one pass
    zero_kernel<<<2048, 256, 0, stream>>>(cnt, (size_t)N_NODES * (1 + CHUNK));
    degree_kernel<<<edgeBlocks, 256, 0, stream>>>(ei, cnt, E);

    gemm_in_ln_kernel<<<GEMM_BLOCKS, BLOCK_THREADS, 0, stream>>>(
        x, W_in, b_in, g_in, be_in, h_a);

    zero_kernel<<<2048, 256, 0, stream>>>(msum, (size_t)N_NODES * HID);
    scatter_add_kernel<<<scatBlocks, 256, 0, stream>>>(ei, h_a, msum, E);
    ongnn_conv_kernel<<<GEMM_BLOCKS, BLOCK_THREADS, 0, stream>>>(
        h_a, msum, cnt, sig, tm_W0, tm_b0, ln_g0, ln_b0, h_b);

    zero_kernel<<<2048, 256, 0, stream>>>(msum, (size_t)N_NODES * HID);
    scatter_add_kernel<<<scatBlocks, 256, 0, stream>>>(ei, h_b, msum, E);
    ongnn_conv_kernel<<<GEMM_BLOCKS, BLOCK_THREADS, 0, stream>>>(
        h_b, msum, cnt, sig, tm_W1, tm_b1, ln_g1, ln_b1, h_a);

    gemm_out_kernel<<<GEMM_BLOCKS, BLOCK_THREADS, 0, stream>>>(h_a, W_out, b_out, out);

    (void)n_in; (void)out_size; (void)ws_size;
}
